// TemporalDecayAttention_74577812127943
// MI455X (gfx1250) — matrix operand order repaired and checked by reading, not compiled, not run on hardware
//
#include <hip/hip_runtime.h>
#include <hip/hip_bf16.h>

typedef _Float16 h16;
typedef __attribute__((ext_vector_type(16))) _Float16 v16h;
typedef __attribute__((ext_vector_type(8)))  _Float16 v8h;
typedef __attribute__((ext_vector_type(8)))  float    v8f;
typedef int v4i __attribute__((vector_size(16)));

#define BB 4
#define TT 2048
#define CC 768
#define HH 12
#define DD 64
#define N3 (3*CC)
#define MM (BB*TT)

#if defined(__has_builtin)
#if __has_builtin(__builtin_amdgcn_global_load_async_to_lds_b128)
#define HAVE_ASYNC_LDS 1
#endif
#endif

static __device__ __forceinline__ v8f vzero8f() {
    v8f z = {0.f,0.f,0.f,0.f,0.f,0.f,0.f,0.f};
    return z;
}
static __device__ __forceinline__ v16h cat8(v8h lo, v8h hi) {
    return __builtin_shufflevector(lo, hi, 0,1,2,3,4,5,6,7,8,9,10,11,12,13,14,15);
}
static __device__ __forceinline__ v8f wmma_f16(v16h a, v16h b, v8f c) {
    // D = A(16x32 f16) x B(32x16 f16) + C(16x16 f32)
    return __builtin_amdgcn_wmma_f32_16x16x32_f16(false, a, false, b, (short)0, c, false, false);
}
#if HAVE_ASYNC_LDS
// async copy of 16 bytes (one b128 per lane) from global to LDS
static __device__ __forceinline__ void async_copy16(const h16* g, h16* l) {
    __builtin_amdgcn_global_load_async_to_lds_b128(
        (__attribute__((address_space(1))) v4i*)(v4i*)const_cast<h16*>(g),
        (__attribute__((address_space(3))) v4i*)(v4i*)l,
        0, 0);
}
#endif
// A-fragment (16x32 f16). rowptr = &src[row*ld + kbase].
// Lanes 0-15 hold K 0..15; lanes 16-31 hold K 16..31 (contiguous).
static __device__ __forceinline__ v16h loadA(const h16* rowptr, int lane) {
    int off = (lane >> 4) * 16;
    v8h lo = *(const v8h*)(rowptr + off);
    v8h hi = *(const v8h*)(rowptr + 8 + off);
    return cat8(lo, hi);
}

// ---------------- conversion / transpose helpers ----------------
__global__ void cvt_f32_f16(const float* __restrict__ src, h16* __restrict__ dst, int n) {
    for (int i = blockIdx.x*blockDim.x + threadIdx.x; i < n; i += gridDim.x*blockDim.x)
        dst[i] = (h16)src[i];
}
// src: [K][N] f32 row-major  ->  dst: [N][K] f16
__global__ void transpose_f32_f16(const float* __restrict__ src, h16* __restrict__ dst, int K, int N) {
    int total = K * N;
    for (int i = blockIdx.x*blockDim.x + threadIdx.x; i < total; i += gridDim.x*blockDim.x) {
        int k = i / N, n = i % N;
        dst[(size_t)n*K + k] = (h16)src[i];
    }
}

// ---------------- QKV GEMM: [8192 x 768] x [768 x 2304] ----------------
__global__ __launch_bounds__(256) void qkv_gemm(const h16* __restrict__ xh,
        const h16* __restrict__ Wat, const float* __restrict__ b_attn,
        h16* __restrict__ qb, h16* __restrict__ kbuf, h16* __restrict__ vtb) {
    int tid = threadIdx.x, lane = tid & 31, wid = tid >> 5;
    int wm = wid >> 1, wn = wid & 1;
    int rbase = blockIdx.y*128 + wm*32;
    int cbase = blockIdx.x*128 + wn*64;
    int l16 = lane & 15, lhi = lane >> 4;

    v8f acc[2][4];
#pragma unroll
    for (int i = 0; i < 2; i++)
#pragma unroll
        for (int j = 0; j < 4; j++) acc[i][j] = vzero8f();

    for (int kk = 0; kk < CC; kk += 32) {
        if (kk + 32 < CC) {
            __builtin_prefetch(xh + (size_t)(rbase + l16)*CC + kk + 32, 0, 1);
            __builtin_prefetch(Wat + (size_t)(cbase + l16)*CC + kk + 32, 0, 1);
        }
        v16h a[2], b[4];
#pragma unroll
        for (int mi = 0; mi < 2; mi++)
            a[mi] = loadA(xh + (size_t)(rbase + mi*16 + l16)*CC + kk, lane);
#pragma unroll
        for (int ni = 0; ni < 4; ni++)
            b[ni] = *(const v16h*)(Wat + (size_t)(cbase + ni*16 + l16)*CC + kk + lhi*16);
#pragma unroll
        for (int mi = 0; mi < 2; mi++)
#pragma unroll
            for (int ni = 0; ni < 4; ni++)
                acc[mi][ni] = wmma_f16(a[mi], b[ni], acc[mi][ni]);
    }

#pragma unroll
    for (int mi = 0; mi < 2; mi++) {
#pragma unroll
        for (int ni = 0; ni < 4; ni++) {
            int n = cbase + ni*16 + l16;
            int sec = n / CC, c = n % CC;
            int h = c >> 6, d = c & 63;
            float bias = b_attn[n];
#pragma unroll
            for (int r = 0; r < 8; r++) {
                int m = rbase + mi*16 + r + 8*lhi;
                int bidx = m >> 11, t = m & (TT-1);
                float val = acc[mi][ni][r] + bias;
                size_t qi = ((size_t)(bidx*HH + h)*TT + t)*DD + d;
                if (sec == 0)      qb[qi]   = (h16)val;
                else if (sec == 1) kbuf[qi] = (h16)val;
                else vtb[((size_t)(bidx*HH + h)*DD + d)*TT + t] = (h16)val;
            }
        }
    }
}

// ---------------- Flash attention with temporal-decay (ALiBi-like) bias ----------------
// grid: (T/64, B*H); 4 waves/block, each wave owns 16 query rows.
__global__ __launch_bounds__(128) void attn_kernel(const h16* __restrict__ qB,
        const h16* __restrict__ kB, const h16* __restrict__ vtB,
        const float* __restrict__ rates, h16* __restrict__ yh) {
    __shared__ alignas(16) h16 pbuf[4][16][32];    // per-wave P-tile scratch
#if HAVE_ASYNC_LDS
    __shared__ alignas(16) h16 ktile[2][32][64];   // double-buffered K tile (32 keys x 64 d)
    __shared__ alignas(16) h16 vtile[2][64][32];   // double-buffered V^T tile (64 d x 32 keys)
#endif
    int tid = threadIdx.x, lane = tid & 31, wid = tid >> 5;
    int l16 = lane & 15, lhi = lane >> 4;
    int bh = blockIdx.y;
    int head = bh % HH, batch = bh / HH;
    int qbase = blockIdx.x*64 + wid*16;

    const h16* qh  = qB  + (size_t)bh*TT*DD;
    const h16* kh  = kB  + (size_t)bh*TT*DD;
    const h16* vth = vtB + (size_t)bh*DD*TT;
    float rate = rates[head];
    const float scale = 0.125f;   // 1/sqrt(64)

    v16h aQ[2];
#pragma unroll
    for (int ks = 0; ks < 2; ks++)
        aQ[ks] = loadA(qh + (size_t)(qbase + l16)*DD + ks*32, lane);

    float mrow[8], lrow[8];
    v8f accO[4];
#pragma unroll
    for (int r = 0; r < 8; r++) { mrow[r] = -1e30f; lrow[r] = 0.f; }
#pragma unroll
    for (int dt = 0; dt < 4; dt++) accO[dt] = vzero8f();

#if HAVE_ASYNC_LDS
    // Block-cooperative, double-buffered async staging of K/V tiles through LDS.
    int kbMax = blockIdx.x*2 + 1;   // = (blockIdx.x*64 + 63) >> 5
    int vd = tid >> 1, vh = tid & 1;
    {   // prologue: issue tile 0 (4 async b128 per thread)
        const h16* gk = kh + tid*16;
        h16* lk = &ktile[0][0][0] + tid*16;
        async_copy16(gk,     lk);
        async_copy16(gk + 8, lk + 8);
        const h16* gv = vth + (size_t)vd*TT + vh*16;
        h16* lv = &vtile[0][vd][vh*16];
        async_copy16(gv,     lv);
        async_copy16(gv + 8, lv + 8);
    }
    for (int kbi = 0; kbi <= kbMax; kbi++) {
        int key0 = kbi*32;
        int cur = kbi & 1, nxt = cur ^ 1;
        bool havenext = kbi < kbMax;
        __syncthreads();   // all waves done consuming buffer 'nxt' from previous round
        if (havenext) {
            int keyn = key0 + 32;
            const h16* gk = kh + (size_t)keyn*DD + tid*16;
            h16* lk = &ktile[nxt][0][0] + tid*16;
            async_copy16(gk,     lk);
            async_copy16(gk + 8, lk + 8);
            const h16* gv = vth + (size_t)vd*TT + keyn + vh*16;
            h16* lv = &vtile[nxt][vd][vh*16];
            async_copy16(gv,     lv);
            async_copy16(gv + 8, lv + 8);
            asm volatile("s_wait_asynccnt 0x4" ::: "memory");  // current tile landed
        } else {
            asm volatile("s_wait_asynccnt 0x0" ::: "memory");
        }
        __syncthreads();   // tile visible to all waves
        if (key0 <= qbase + 15) {
            v8f sacc[2];
#pragma unroll
            for (int nt = 0; nt < 2; nt++) {
                v8f c = vzero8f();
#pragma unroll
                for (int ks = 0; ks < 2; ks++) {
                    v16h bK = *(const v16h*)(&ktile[cur][nt*16 + l16][ks*32 + lhi*16]);
                    c = wmma_f16(aQ[ks], bK, c);
                }
                sacc[nt] = c;
            }
            float p0a[8], p1a[8];
#pragma unroll
            for (int r = 0; r < 8; r++) {
                int i = qbase + r + 8*lhi;
                int j0 = key0 + l16, j1 = key0 + 16 + l16;
                float s0 = sacc[0][r]*scale, s1 = sacc[1][r]*scale;
                s0 = (j0 > i) ? -1e30f : (s0 - rate*(float)(i - j0));
                s1 = (j1 > i) ? -1e30f : (s1 - rate*(float)(i - j1));
                float mx = fmaxf(s0, s1);
#pragma unroll
                for (int msk = 8; msk >= 1; msk >>= 1) mx = fmaxf(mx, __shfl_xor(mx, msk, 32));
                float mnew = fmaxf(mrow[r], mx);
                float corr = __expf(mrow[r] - mnew);
                float p0 = __expf(s0 - mnew), p1 = __expf(s1 - mnew);
                float rs = p0 + p1;
#pragma unroll
                for (int msk = 8; msk >= 1; msk >>= 1) rs += __shfl_xor(rs, msk, 32);
                lrow[r] = lrow[r]*corr + rs;
                mrow[r] = mnew;
#pragma unroll
                for (int dt = 0; dt < 4; dt++) accO[dt][r] = accO[dt][r] * corr;
                p0a[r] = p0; p1a[r] = p1;
            }
#pragma unroll
            for (int r = 0; r < 8; r++) {
                pbuf[wid][r + 8*lhi][l16]      = (h16)p0a[r];
                pbuf[wid][r + 8*lhi][16 + l16] = (h16)p1a[r];
            }
            asm volatile("" ::: "memory");
            v8h plo = *(const v8h*)(&pbuf[wid][l16][lhi*16]);
            v8h phi = *(const v8h*)(&pbuf[wid][l16][8 + lhi*16]);
            v16h aP = cat8(plo, phi);
            asm volatile("" ::: "memory");
#pragma unroll
            for (int dt = 0; dt < 4; dt++) {
                v16h bV = *(const v16h*)(&vtile[cur][dt*16 + l16][lhi*16]);
                accO[dt] = wmma_f16(aP, bV, accO[dt]);
            }
        }
    }
#else
    // Fallback: direct-global fragment loads, per-wave causal trip count.
    int kbTop = (qbase + 15) >> 5;
    for (int kbi = 0; kbi <= kbTop; kbi++) {
        int key0 = kbi*32;
        v8f sacc[2];
#pragma unroll
        for (int nt = 0; nt < 2; nt++) {
            v8f c = vzero8f();
#pragma unroll
            for (int ks = 0; ks < 2; ks++) {
                v16h bK = *(const v16h*)(kh + (size_t)(key0 + nt*16 + l16)*DD + ks*32 + lhi*16);
                c = wmma_f16(aQ[ks], bK, c);
            }
            sacc[nt] = c;
        }
        float p0a[8], p1a[8];
#pragma unroll
        for (int r = 0; r < 8; r++) {
            int i = qbase + r + 8*lhi;
            int j0 = key0 + l16, j1 = key0 + 16 + l16;
            float s0 = sacc[0][r]*scale, s1 = sacc[1][r]*scale;
            s0 = (j0 > i) ? -1e30f : (s0 - rate*(float)(i - j0));
            s1 = (j1 > i) ? -1e30f : (s1 - rate*(float)(i - j1));
            float mx = fmaxf(s0, s1);
#pragma unroll
            for (int msk = 8; msk >= 1; msk >>= 1) mx = fmaxf(mx, __shfl_xor(mx, msk, 32));
            float mnew = fmaxf(mrow[r], mx);
            float corr = __expf(mrow[r] - mnew);
            float p0 = __expf(s0 - mnew), p1 = __expf(s1 - mnew);
            float rs = p0 + p1;
#pragma unroll
            for (int msk = 8; msk >= 1; msk >>= 1) rs += __shfl_xor(rs, msk, 32);
            lrow[r] = lrow[r]*corr + rs;
            mrow[r] = mnew;
#pragma unroll
            for (int dt = 0; dt < 4; dt++) accO[dt][r] = accO[dt][r] * corr;
            p0a[r] = p0; p1a[r] = p1;
        }
#pragma unroll
        for (int r = 0; r < 8; r++) {
            pbuf[wid][r + 8*lhi][l16]      = (h16)p0a[r];
            pbuf[wid][r + 8*lhi][16 + l16] = (h16)p1a[r];
        }
        asm volatile("" ::: "memory");
        v8h plo = *(const v8h*)(&pbuf[wid][l16][lhi*16]);
        v8h phi = *(const v8h*)(&pbuf[wid][l16][8 + lhi*16]);
        v16h aP = cat8(plo, phi);
        asm volatile("" ::: "memory");
#pragma unroll
        for (int dt = 0; dt < 4; dt++) {
            v16h bV = *(const v16h*)(vth + (size_t)(dt*16 + l16)*TT + key0 + lhi*16);
            accO[dt] = wmma_f16(aP, bV, accO[dt]);
        }
        asm volatile("" ::: "memory");
    }
#endif

    float inv[8];
#pragma unroll
    for (int r = 0; r < 8; r++) inv[r] = 1.0f / (lrow[r] + 1e-8f);
#pragma unroll
    for (int dt = 0; dt < 4; dt++) {
        int col = head*DD + dt*16 + l16;
#pragma unroll
        for (int r = 0; r < 8; r++) {
            int t = qbase + r + 8*lhi;
            yh[((size_t)batch*TT + t)*CC + col] = (h16)(accO[dt][r]*inv[r]);
        }
    }
}

// ---------------- Output projection: [8192 x 768] x [768 x 768] -> fp32 ----------------
__global__ __launch_bounds__(256) void proj_gemm(const h16* __restrict__ yh,
        const h16* __restrict__ Wpt, const float* __restrict__ b_proj, float* __restrict__ out) {
    int tid = threadIdx.x, lane = tid & 31, wid = tid >> 5;
    int wm = wid >> 1, wn = wid & 1;
    int rbase = blockIdx.y*128 + wm*32;
    int cbase = blockIdx.x*128 + wn*64;
    int l16 = lane & 15, lhi = lane >> 4;

    v8f acc[2][4];
#pragma unroll
    for (int i = 0; i < 2; i++)
#pragma unroll
        for (int j = 0; j < 4; j++) acc[i][j] = vzero8f();

    for (int kk = 0; kk < CC; kk += 32) {
        if (kk + 32 < CC) {
            __builtin_prefetch(yh + (size_t)(rbase + l16)*CC + kk + 32, 0, 1);
            __builtin_prefetch(Wpt + (size_t)(cbase + l16)*CC + kk + 32, 0, 1);
        }
        v16h a[2], b[4];
#pragma unroll
        for (int mi = 0; mi < 2; mi++)
            a[mi] = loadA(yh + (size_t)(rbase + mi*16 + l16)*CC + kk, lane);
#pragma unroll
        for (int ni = 0; ni < 4; ni++)
            b[ni] = *(const v16h*)(Wpt + (size_t)(cbase + ni*16 + l16)*CC + kk + lhi*16);
#pragma unroll
        for (int mi = 0; mi < 2; mi++)
#pragma unroll
            for (int ni = 0; ni < 4; ni++)
                acc[mi][ni] = wmma_f16(a[mi], b[ni], acc[mi][ni]);
    }
#pragma unroll
    for (int mi = 0; mi < 2; mi++) {
#pragma unroll
        for (int ni = 0; ni < 4; ni++) {
            int n = cbase + ni*16 + l16;
            float bias = b_proj[n];
#pragma unroll
            for (int r = 0; r < 8; r++) {
                int m = rbase + mi*16 + r + 8*lhi;
                out[(size_t)m*CC + n] = acc[mi][ni][r] + bias;
            }
        }
    }
}

extern "C" void kernel_launch(void* const* d_in, const int* in_sizes, int n_in,
                              void* d_out, int out_size, void* d_ws, size_t ws_size,
                              hipStream_t stream) {
    const float* x       = (const float*)d_in[0];
    const float* W_attn  = (const float*)d_in[1];
    const float* b_attn  = (const float*)d_in[2];
    const float* W_proj  = (const float*)d_in[3];
    const float* b_proj  = (const float*)d_in[4];
    const float* rates   = (const float*)d_in[5];
    (void)in_sizes; (void)n_in; (void)out_size; (void)ws_size;

    h16* ws  = (h16*)d_ws;
    h16* xh  = ws;
    h16* Wat = xh  + (size_t)MM*CC;
    h16* Wpt = Wat + (size_t)N3*CC;
    h16* qb  = Wpt + (size_t)CC*CC;
    h16* kb  = qb  + (size_t)BB*HH*TT*DD;
    h16* vtb = kb  + (size_t)BB*HH*TT*DD;
    h16* yhb = vtb + (size_t)BB*HH*TT*DD;

    cvt_f32_f16<<<4096, 256, 0, stream>>>(x, xh, MM*CC);
    transpose_f32_f16<<<2048, 256, 0, stream>>>(W_attn, Wat, CC, N3);
    transpose_f32_f16<<<1024, 256, 0, stream>>>(W_proj, Wpt, CC, CC);

    qkv_gemm<<<dim3(N3/128, MM/128), 256, 0, stream>>>(xh, Wat, b_attn, qb, kb, vtb);
    attn_kernel<<<dim3(TT/64, BB*HH), 128, 0, stream>>>(qb, kb, vtb, rates, yhb);
    proj_gemm<<<dim3(CC/128, MM/128), 256, 0, stream>>>(yhb, Wpt, b_proj, (float*)d_out);
}
